// VQ_EWMA_kmeans_231928234657
// MI455X (gfx1250) — matrix operand order repaired and checked by reading, not compiled, not run on hardware
//
#include <hip/hip_runtime.h>
#include <hip/hip_bf16.h>

// ---------------------------------------------------------------------------
// VQ EWMA k-means for MI455X (gfx1250, wave32, WMMA + async-to-LDS pipeline).
// Score GEMM done as split-bf16 (hi+lo) on v_wmma_f32_16x16x32_bf16:
//   x.v ~= xh.vh + xl.vh + xh.vl   (error ~2^-16 relative, ~fp32 argmin)
// B tiles (vq hi/lo planes) are staged into LDS once per block with
// global_load_async_to_lds_b128 in a 4-deep pipeline (ASYNCcnt), so the
// 12 WMMAs/tile overlap the L2 fetch of tile kt+3 and L2 traffic drops 4x.
// ---------------------------------------------------------------------------

#define GAMMA_F 0.99f
#define OMG_F   ((float)(1.0 - 0.99))   // matches reference (1.0 - GAMMA)

#define DEPTH    4                       // pipeline depth (power of 2)
#define HPITCH   144                     // LDS row pitch: 128B data + 16B pad
#define PLANE    (16 * HPITCH)           // one bf16 plane (16 rows)   = 2304 B
#define BUFBYTES (2 * PLANE)             // hi + lo planes per tile    = 4608 B
#define ASYNC_WAIT ((DEPTH - 2) * 2)     // 2 chunks/thread/batch, keep 2 batches in flight

typedef __attribute__((ext_vector_type(16))) __bf16        v16bf;
typedef __attribute__((ext_vector_type(8)))  float         v8f;
typedef __attribute__((ext_vector_type(8)))  unsigned int  v8u;

union BU { v8u u; v16bf b; };

__device__ __forceinline__ unsigned int f2bf(float f) {
  // round-to-nearest-even fp32 -> bf16 bits
  unsigned int u = __float_as_uint(f);
  return (u + 0x7FFFu + ((u >> 16) & 1u)) >> 16;
}
__device__ __forceinline__ float bf2f(unsigned int h) {
  return __uint_as_float(h << 16);
}

__device__ __forceinline__ v8f wmma_bf16(v8u a, v8u b, v8f c) {
  BU ua; ua.u = a;
  BU ub; ub.u = b;
  return __builtin_amdgcn_wmma_f32_16x16x32_bf16(false, ua.b, false, ub.b,
                                                 (short)0, c, false, false);
}

// issue one 16B async global->LDS chunk (GVS addressing: saddr + 32b voffset)
__device__ __forceinline__ void async_b128(unsigned lds_addr, unsigned goff,
                                           unsigned long long sbase) {
  asm volatile("global_load_async_to_lds_b128 %0, %1, %2"
               :: "v"(lds_addr), "v"(goff), "s"(sbase) : "memory");
}

// ---------------------------------------------------------------------------
// Kernel 1: split vq (fp32 [K,64]) into bf16 hi/lo planes + fp32 row norms.
// ---------------------------------------------------------------------------
__global__ __launch_bounds__(256) void vq_prep_kernel(
    const float* __restrict__ vq,
    unsigned short* __restrict__ vq_hi,
    unsigned short* __restrict__ vq_lo,
    float* __restrict__ vqn) {
  const int lane = threadIdx.x & 31;
  const int row  = blockIdx.x * 8 + (threadIdx.x >> 5);
  const float* p = vq + (size_t)row * 64;
  float f0 = p[lane], f1 = p[lane + 32];
  unsigned int h0 = f2bf(f0), h1 = f2bf(f1);
  vq_hi[(size_t)row * 64 + lane]      = (unsigned short)h0;
  vq_hi[(size_t)row * 64 + lane + 32] = (unsigned short)h1;
  vq_lo[(size_t)row * 64 + lane]      = (unsigned short)f2bf(f0 - bf2f(h0));
  vq_lo[(size_t)row * 64 + lane + 32] = (unsigned short)f2bf(f1 - bf2f(h1));
  float s = f0 * f0 + f1 * f1;
#pragma unroll
  for (int d = 16; d >= 1; d >>= 1) s += __shfl_xor(s, d, 32);
  if (lane == 0) vqn[row] = s;
}

// ---------------------------------------------------------------------------
// Kernel 2: seed accumulators: new_sum = g*ewma_sum, new_n = g*ewma_n.
// ---------------------------------------------------------------------------
__global__ __launch_bounds__(256) void vq_init_kernel(
    const float* __restrict__ esum, const float* __restrict__ en,
    float* __restrict__ out_sum, float* __restrict__ out_n, int K) {
  const int i = blockIdx.x * 256 + threadIdx.x;
  out_sum[i] = GAMMA_F * esum[i];
  if (i < K) out_n[i] = GAMMA_F * en[i];
}

// ---------------------------------------------------------------------------
// Kernel 3: one wave = 32 rows of x. A fragments (split-bf16) built once and
// pinned in VGPRs. K loop consumes LDS-staged B tiles (async pipeline).
// ---------------------------------------------------------------------------
__global__ __launch_bounds__(128) void vq_assign_kernel(
    const float* __restrict__ x, const float* __restrict__ vq,
    const unsigned short* __restrict__ vq_hi,
    const unsigned short* __restrict__ vq_lo,
    const float* __restrict__ vqn, int ktiles,
    float* __restrict__ out_q, float* __restrict__ out_sum,
    float* __restrict__ out_n) {
  __shared__ alignas(16) unsigned char smem[DEPTH][BUFBYTES];

  const int t       = threadIdx.x;
  const int lane    = t & 31;
  const int wave    = t >> 5;
  const int wtile   = blockIdx.x * 4 + wave;
  const int rowbase = wtile * 32;
  const int nloc    = lane & 15;
  const int khalfA  = (lane >> 4) * 8;    // A layout: upper half-wave K+8
  const int bcol    = (lane >> 4) * 16;   // B layout: upper half-wave K+8 (bytes)

  const unsigned lds0 = (unsigned)(uintptr_t)&smem[0][0];
  const int cr = t >> 3;                  // staging: row 0..15
  const int cc = (t & 7) * 16;            // staging: 16B column within row
  const unsigned long long hbase = (unsigned long long)(uintptr_t)vq_hi;
  const unsigned long long lbase = (unsigned long long)(uintptr_t)vq_lo;

  // --- A fragments: 2 M-tiles x 2 K-chunks, hi & lo planes
  v8u Ah[2][2], Al[2][2];
#pragma unroll
  for (int mt = 0; mt < 2; ++mt) {
    const float* px = x + (size_t)(rowbase + mt * 16 + nloc) * 64;
#pragma unroll
    for (int kc = 0; kc < 2; ++kc) {
      const int b0 = kc * 32 + khalfA;
      float f[16];
      const float4* q0 = reinterpret_cast<const float4*>(px + b0);
      const float4* q1 = reinterpret_cast<const float4*>(px + b0 + 16);
      float4 a0 = q0[0], a1 = q0[1], b1 = q1[0], b2 = q1[1];
      f[0]=a0.x; f[1]=a0.y; f[2]=a0.z; f[3]=a0.w;
      f[4]=a1.x; f[5]=a1.y; f[6]=a1.z; f[7]=a1.w;
      f[8]=b1.x; f[9]=b1.y; f[10]=b1.z; f[11]=b1.w;
      f[12]=b2.x; f[13]=b2.y; f[14]=b2.z; f[15]=b2.w;
#pragma unroll
      for (int v = 0; v < 8; ++v) {
        unsigned int h0 = f2bf(f[2*v]), h1 = f2bf(f[2*v+1]);
        Ah[mt][kc][v] = h0 | (h1 << 16);
        unsigned int l0 = f2bf(f[2*v]   - bf2f(h0));
        unsigned int l1 = f2bf(f[2*v+1] - bf2f(h1));
        Al[mt][kc][v] = l0 | (l1 << 16);
      }
    }
  }

  // --- prime async pipeline: batches for tiles 0..DEPTH-2
  for (int p = 0; p < DEPTH - 1 && p < ktiles; ++p) {
    unsigned ldsh = lds0 + (unsigned)p * BUFBYTES + (unsigned)(cr * HPITCH + cc);
    unsigned goff = (unsigned)p * 2048u + (unsigned)t * 16u;
    async_b128(ldsh,         goff, hbase);
    async_b128(ldsh + PLANE, goff, lbase);
  }
  asm volatile("s_wait_asynccnt %0" :: "n"(ASYNC_WAIT) : "memory");
  __syncthreads();

  float minv[2][8];
  int   mini[2][8];
#pragma unroll
  for (int mt = 0; mt < 2; ++mt)
#pragma unroll
    for (int j = 0; j < 8; ++j) { minv[mt][j] = __builtin_inff(); mini[mt][j] = 0; }

  float vn_next = vqn[nloc];   // tile 0 norms

  for (int kt = 0; kt < ktiles; ++kt) {
    // prefetch batch kt+DEPTH-1 (wrap at end: redundant but keeps waits uniform)
    {
      int pf = kt + DEPTH - 1;
      if (pf >= ktiles) pf -= ktiles;
      unsigned slot = (unsigned)((kt + DEPTH - 1) & (DEPTH - 1));
      unsigned ldsh = lds0 + slot * BUFBYTES + (unsigned)(cr * HPITCH + cc);
      unsigned goff = (unsigned)pf * 2048u + (unsigned)t * 16u;
      async_b128(ldsh,         goff, hbase);
      async_b128(ldsh + PLANE, goff, lbase);
    }

    const float vn = vn_next;                       // norms for tile kt
    int nn = (kt + 1 < ktiles) ? (kt + 1) : 0;      // register-prefetch next norms
    vn_next = vqn[nn * 16 + nloc];

    // --- B fragments from LDS (ds_load_b128, conflict-free via 144B pitch)
    const unsigned char* bb = &smem[kt & (DEPTH - 1)][0] + nloc * HPITCH + bcol;
    v8u Bh[2], Bl[2];
#pragma unroll
    for (int kc = 0; kc < 2; ++kc) {
      const uint4* ph = reinterpret_cast<const uint4*>(bb + kc * 64);
      const uint4* pl = reinterpret_cast<const uint4*>(bb + PLANE + kc * 64);
      uint4 h0 = ph[0], h1 = ph[2], l0 = pl[0], l1 = pl[2];
      Bh[kc][0]=h0.x; Bh[kc][1]=h0.y; Bh[kc][2]=h0.z; Bh[kc][3]=h0.w;
      Bh[kc][4]=h1.x; Bh[kc][5]=h1.y; Bh[kc][6]=h1.z; Bh[kc][7]=h1.w;
      Bl[kc][0]=l0.x; Bl[kc][1]=l0.y; Bl[kc][2]=l0.z; Bl[kc][3]=l0.w;
      Bl[kc][4]=l1.x; Bl[kc][5]=l1.y; Bl[kc][6]=l1.z; Bl[kc][7]=l1.w;
    }

    const int n = kt * 16 + nloc;
#pragma unroll
    for (int mt = 0; mt < 2; ++mt) {
      v8f acc = {0.f,0.f,0.f,0.f,0.f,0.f,0.f,0.f};
      acc = wmma_bf16(Ah[mt][0], Bh[0], acc);
      acc = wmma_bf16(Al[mt][0], Bh[0], acc);
      acc = wmma_bf16(Ah[mt][0], Bl[0], acc);
      acc = wmma_bf16(Ah[mt][1], Bh[1], acc);
      acc = wmma_bf16(Al[mt][1], Bh[1], acc);
      acc = wmma_bf16(Ah[mt][1], Bl[1], acc);
#pragma unroll
      for (int j = 0; j < 8; ++j) {
        float s = fmaf(-2.0f, acc[j], vn);          // ||v||^2 - 2 x.v
        if (s < minv[mt][j]) { minv[mt][j] = s; mini[mt][j] = n; }
      }
    }

    // publish tile kt+1 (oldest in-flight batch must have landed)
    asm volatile("s_wait_asynccnt %0" :: "n"(ASYNC_WAIT) : "memory");
    __syncthreads();
  }

  // --- min-reduce across the 16 lanes sharing one M row (tie -> lower index)
#pragma unroll
  for (int mt = 0; mt < 2; ++mt)
#pragma unroll
    for (int j = 0; j < 8; ++j)
#pragma unroll
      for (int d = 1; d < 16; d <<= 1) {
        float ov = __shfl_xor(minv[mt][j], d, 32);
        int   oi = __shfl_xor(mini[mt][j], d, 32);
        if (ov < minv[mt][j] || (ov == minv[mt][j] && oi < mini[mt][j])) {
          minv[mt][j] = ov; mini[mt][j] = oi;
        }
      }

  // --- outputs: gather vq[idx] into quantized, scatter (1-g)*x into new_sum
  const float om = OMG_F;
#pragma unroll
  for (int mt = 0; mt < 2; ++mt) {
#pragma unroll
    for (int rr = 0; rr < 16; ++rr) {
      // lane 0 holds M=0..7, lane 16 holds M=8..15 (C-matrix layout)
      const int idx = __shfl(mini[mt][rr & 7], (rr >> 3) * 16, 32);
      const int row = rowbase + mt * 16 + rr;
      const float* vrow = vq + (size_t)idx * 64;
      const float* xrow = x + (size_t)row * 64;
      out_q[(size_t)row * 64 + lane]      = vrow[lane];
      out_q[(size_t)row * 64 + lane + 32] = vrow[lane + 32];
      atomicAdd(&out_sum[(size_t)idx * 64 + lane],      om * xrow[lane]);
      atomicAdd(&out_sum[(size_t)idx * 64 + lane + 32], om * xrow[lane + 32]);
      if (lane == 0) atomicAdd(&out_n[idx], om);
    }
  }
}

// ---------------------------------------------------------------------------
// Kernel 4: new_vq = new_sum / new_n.
// ---------------------------------------------------------------------------
__global__ __launch_bounds__(256) void vq_final_kernel(
    const float* __restrict__ out_sum, const float* __restrict__ out_n,
    float* __restrict__ out_vq) {
  const int i = blockIdx.x * 256 + threadIdx.x;
  out_vq[i] = out_sum[i] / out_n[i >> 6];
}

// ---------------------------------------------------------------------------
extern "C" void kernel_launch(void* const* d_in, const int* in_sizes, int n_in,
                              void* d_out, int out_size, void* d_ws, size_t ws_size,
                              hipStream_t stream) {
  const float* x    = (const float*)d_in[0];  // [N,64]
  const float* vq   = (const float*)d_in[1];  // [K,64]
  const float* esum = (const float*)d_in[2];  // [K,64]
  const float* en   = (const float*)d_in[3];  // [K]
  const int D = 64;
  const int N = in_sizes[0] / D;              // 65536
  const int K = in_sizes[1] / D;              // 4096

  // workspace: bf16 hi plane | bf16 lo plane | fp32 vq norms  (~1.06 MB)
  unsigned short* vq_hi = (unsigned short*)d_ws;
  unsigned short* vq_lo = vq_hi + (size_t)K * D;
  float*          vqn   = (float*)(vq_lo + (size_t)K * D);

  // d_out: quantized [N*D] | new_vq [K*D] | new_sum [K*D] | new_n [K]
  float* out_q   = (float*)d_out;
  float* out_vq  = out_q   + (size_t)N * D;
  float* out_sum = out_vq  + (size_t)K * D;
  float* out_n   = out_sum + (size_t)K * D;

  vq_prep_kernel<<<K / 8, 256, 0, stream>>>(vq, vq_hi, vq_lo, vqn);
  vq_init_kernel<<<(K * D) / 256, 256, 0, stream>>>(esum, en, out_sum, out_n, K);
  vq_assign_kernel<<<N / 128, 128, 0, stream>>>(x, vq, vq_hi, vq_lo, vqn, K / 16,
                                                out_q, out_sum, out_n);
  vq_final_kernel<<<(K * D) / 256, 256, 0, stream>>>(out_sum, out_n, out_vq);
}
